// InvariantPointCrossAttention_56014963475092
// MI455X (gfx1250) — matrix operand order repaired and checked by reading, not compiled, not run on hardware
//
#include <hip/hip_runtime.h>
#include <hip/hip_bf16.h>
#include <math.h>

// ---------------------------------------------------------------------------
// Invariant Point Cross Attention for MI455X (gfx1250), wave32 + WMMA bf16
// + Tensor Data Mover staging of the pair tensor.
// ---------------------------------------------------------------------------

typedef __attribute__((ext_vector_type(16))) __bf16 v16bf;
typedef __attribute__((ext_vector_type(8)))  float  v8f;
typedef __attribute__((ext_vector_type(4)))  int    s4i;
typedef __attribute__((ext_vector_type(8)))  int    s8i;

#define Bq 2
#define Nn 1024
#define Mm 1024
#define Dd 768
#define Hh 12
#define PQn 4
#define PVn 8
#define Cc 24
#define DHn 64
#define FEAT 1440   // H*(C + DH + PV*4) = 12*120

// --------------------------- workspace layout (floats) ----------------------
// total ~50.6M floats (~203 MB) — scores f32 (100MB) + probs bf16 (50MB) dominate
__host__ __device__ constexpr size_t SZ_QKV   = (size_t)Bq * Nn * Dd;
__host__ __device__ constexpr size_t SZ_QP    = (size_t)Bq * Nn * Hh * PQn * 3;
__host__ __device__ constexpr size_t SZ_VPP   = (size_t)Bq * Mm * Hh * PVn * 3;
__host__ __device__ constexpr size_t SZ_APTS  = (size_t)Bq * Hh * Nn * 12;
__host__ __device__ constexpr size_t SZ_CPTS  = (size_t)Bq * Hh * Mm * 24;
__host__ __device__ constexpr size_t SZ_SQ    = (size_t)Bq * Hh * Nn;
__host__ __device__ constexpr size_t SZ_SC    = (size_t)Bq * Hh * Nn * Mm;
__host__ __device__ constexpr size_t SZ_CTX   = (size_t)Bq * Nn * Hh * DHn;
__host__ __device__ constexpr size_t SZ_POG   = (size_t)Bq * Hh * Nn * 24;
__host__ __device__ constexpr size_t SZ_POUT  = (size_t)Bq * Nn * Hh * Cc;
__host__ __device__ constexpr size_t SZ_FE    = (size_t)Bq * Nn * FEAT;

__host__ __device__ constexpr size_t OFF_Q     = 0;
__host__ __device__ constexpr size_t OFF_K     = OFF_Q    + SZ_QKV;
__host__ __device__ constexpr size_t OFF_V     = OFF_K    + SZ_QKV;
__host__ __device__ constexpr size_t OFF_QP    = OFF_V    + SZ_QKV;
__host__ __device__ constexpr size_t OFF_KP    = OFF_QP   + SZ_QP;
__host__ __device__ constexpr size_t OFF_VP    = OFF_KP   + SZ_QP;
__host__ __device__ constexpr size_t OFF_APTS  = OFF_VP   + SZ_VPP;
__host__ __device__ constexpr size_t OFF_BPTS  = OFF_APTS + SZ_APTS;
__host__ __device__ constexpr size_t OFF_CPTS  = OFF_BPTS + SZ_APTS;
__host__ __device__ constexpr size_t OFF_ASQ   = OFF_CPTS + SZ_CPTS;
__host__ __device__ constexpr size_t OFF_BSQ   = OFF_ASQ  + SZ_SQ;
__host__ __device__ constexpr size_t OFF_SC    = OFF_BSQ  + SZ_SQ;
__host__ __device__ constexpr size_t OFF_CTX   = OFF_SC   + SZ_SC;
__host__ __device__ constexpr size_t OFF_POG   = OFF_CTX  + SZ_CTX;
__host__ __device__ constexpr size_t OFF_POUT  = OFF_POG  + SZ_POG;
__host__ __device__ constexpr size_t OFF_FE    = OFF_POUT + SZ_POUT;
__host__ __device__ constexpr size_t OFF_PROBS = OFF_FE   + SZ_FE;   // bf16 region, SZ_SC elems

// ----------------------- fragment K-index helpers (ISA 7.12.2) --------------
__device__ __forceinline__ int a_kidx(int i, int j, int half) {
  return (i < 4) ? (8 * half + 2 * i + j) : (16 + 8 * half + 2 * (i - 4) + j);
}
__device__ __forceinline__ int b_kidx(int v, int j, int half) {
  return (v < 4) ? (8 * half + 2 * v + j) : (16 + 8 * half + 2 * (v - 4) + j);
}

// ----------------------- Tensor Data Mover (ISA ch.8, D# groups) ------------
// 3D tile load: LDS[z][y][x] <- GLOBAL[base + dsz*(x + y*stride0 + z*stride1)]
// Issue from ONE wave only (TDM ignores EXEC); follow with s_wait_tensorcnt.
__device__ __forceinline__ void tdm_load_to_lds(
    unsigned lds_byte, const void* gptr, unsigned dsz_log,
    unsigned tile0, unsigned tile1, unsigned tile2,
    unsigned long long stride0, unsigned long long stride1)
{
  const unsigned long long ga    = (unsigned long long)(size_t)gptr;
  const unsigned long long tdim0 = 1ull << 20;   // generous (tile is in-bounds)
  const unsigned long long tdim1 = 1ull << 20;
  s4i g0;
  g0[0] = 1;                                               // count=1, user mode
  g0[1] = (int)lds_byte;                                   // lds_addr
  g0[2] = (int)(ga & 0xffffffffull);                       // global_addr[31:0]
  g0[3] = (int)((ga >> 32) & 0x01ffffffull) | (int)(2u << 30); // addr[56:32] | type=2
  s8i g1;
  g1[0] = (int)(dsz_log << 16);                            // wg_mask=0 | data_size
  g1[1] = (int)((tdim0 & 0xffffull) << 16);                // tensor_dim0[15:0]
  g1[2] = (int)(((tdim0 >> 16) & 0xffffull) | ((tdim1 & 0xffffull) << 16));
  g1[3] = (int)(((tdim1 >> 16) & 0xffffull) | ((unsigned long long)tile0 << 16));
  g1[4] = (int)(tile1 | (tile2 << 16));                    // tile_dim1 | tile_dim2
  g1[5] = (int)(stride0 & 0xffffffffull);                  // dim0_stride[31:0]
  g1[6] = (int)(((stride0 >> 32) & 0xffffull) | ((stride1 & 0xffffull) << 16));
  g1[7] = (int)((stride1 >> 16) & 0xffffffffull);          // dim1_stride[47:16]
  s4i g2;
  g2[0] = (int)tile2;   // tensor_dim2 (OOB extent)
  g2[1] = 0;            // tensor_dim3
  g2[2] = 0;            // tensor_dim2_stride lo
  g2[3] = 0;            // stride hi | tile_dim3=0
  s4i g3 = {0, 0, 0, 0};
  asm volatile("tensor_load_to_lds %0, %1, %2, %3"
               :: "s"(g0), "s"(g1), "s"(g2), "s"(g3)
               : "memory");
}

// ===========================================================================
// Kernel 1: generic tiled GEMM  out[rows x Nout] = A[rows x K] * W[K x Nout]
//   Requires rows%64==0, K%32==0, Nout%4==0 (all call sites satisfy this).
// ===========================================================================
#define GT_KT 32
#define ASTR 36   // 64 x (32+4) floats, 16B aligned rows, conflict-free cols
#define WSTR 68   // 32 x (64+4) floats

__global__ __launch_bounds__(256)
void gemm_bf16_kernel(const float* __restrict__ A, const float* __restrict__ W,
                      const float* __restrict__ bias, float* __restrict__ out,
                      int rows, int K, int Nout)
{
  __shared__ float As[64 * ASTR];
  __shared__ float Ws[GT_KT * WSTR];

  const int tid  = threadIdx.x;
  const int lane = tid & 31;
  const int wid  = tid >> 5;       // 0..7
  const int rsub = wid & 3;        // 16-row strip within 64
  const int cg   = wid >> 2;       // 0..1 -> 32-wide col group
  const int row0 = blockIdx.y * 64;
  const int col0 = blockIdx.x * 64;
  const int m16  = lane & 15;
  const int half = lane >> 4;

  // staging decomposition (fixed per thread, 2 float4 each for A and W)
  const int ar  = tid >> 3;             // 0..31  (+32 for second vector)
  const int ac4 = (tid & 7) << 2;       // 0,4,..,28
  const int wk  = tid >> 4;             // 0..15  (+16 for second vector)
  const int wc4 = (tid & 15) << 2;      // 0,4,..,60
  const int wgc = col0 + wc4;
  const bool wok = (wgc < Nout);

  const float* gA = A + (size_t)(row0 + ar) * K + ac4;           // advances +32/iter
  const float* gW = W + (size_t)wk * Nout + wgc;                 // advances +32*Nout/iter

  v8f cacc[2] = {};

  for (int kt = 0; kt < K; kt += GT_KT) {
    *(float4*)(&As[ar * ASTR + ac4])        = *(const float4*)(gA);
    *(float4*)(&As[(ar + 32) * ASTR + ac4]) = *(const float4*)(gA + (size_t)32 * K);
    float4 w0 = {0.f, 0.f, 0.f, 0.f}, w1 = {0.f, 0.f, 0.f, 0.f};
    if (wok) {
      w0 = *(const float4*)(gW);
      w1 = *(const float4*)(gW + (size_t)16 * Nout);
    }
    *(float4*)(&Ws[wk * WSTR + wc4])        = w0;
    *(float4*)(&Ws[(wk + 16) * WSTR + wc4]) = w1;
    gA += GT_KT;
    gW += (size_t)GT_KT * Nout;
    __syncthreads();

    v16bf af;
#pragma unroll
    for (int i = 0; i < 8; ++i)
#pragma unroll
      for (int j = 0; j < 2; ++j)
        af[2 * i + j] = (__bf16)As[(rsub * 16 + m16) * ASTR + a_kidx(i, j, half)];

#pragma unroll
    for (int s = 0; s < 2; ++s) {
      v16bf bf;
      int cbase = cg * 32 + s * 16;
#pragma unroll
      for (int v = 0; v < 8; ++v)
#pragma unroll
        for (int j = 0; j < 2; ++j)
          bf[2 * v + j] = (__bf16)Ws[b_kidx(v, j, half) * WSTR + cbase + m16];
      cacc[s] = __builtin_amdgcn_wmma_f32_16x16x32_bf16(
          false, af, false, bf, (short)0, cacc[s], false, false);
    }
    __syncthreads();
  }

#pragma unroll
  for (int v = 0; v < 8; ++v) {
    int r = row0 + rsub * 16 + v + 8 * half;
#pragma unroll
    for (int s = 0; s < 2; ++s) {
      int c = col0 + cg * 32 + s * 16 + m16;
      if (c < Nout) {
        float val = cacc[s][v];
        if (bias) val += bias[c];
        out[(size_t)r * Nout + c] = val;
      }
    }
  }
}

// ===========================================================================
// Kernel 2: rigid-rotate point projections into global frame + squared norms
// ===========================================================================
__global__ __launch_bounds__(256)
void rotate_points_kernel(const float* __restrict__ qp, const float* __restrict__ kp,
                          const float* __restrict__ vp,
                          const float* __restrict__ R,  const float* __restrict__ t,
                          const float* __restrict__ Re, const float* __restrict__ te,
                          float* __restrict__ aPts, float* __restrict__ bPts,
                          float* __restrict__ cPts, float* __restrict__ aSq,
                          float* __restrict__ bSq)
{
  int idx = blockIdx.x * blockDim.x + threadIdx.x;   // over B*N*H (N==M)
  if (idx >= Bq * Nn * Hh) return;
  int h = idx % Hh;
  int n = (idx / Hh) % Nn;
  int b = idx / (Hh * Nn);

  { // a-side (query points)
    const float* Rn = R + ((size_t)(b * Nn + n)) * 9;
    const float* tn = t + ((size_t)(b * Nn + n)) * 3;
    float sq = 0.f;
    for (int p = 0; p < PQn; ++p) {
      const float* s = qp + ((size_t)(b * Nn + n)) * (Hh * PQn * 3) + (h * PQn + p) * 3;
      float x = s[0], y = s[1], z = s[2];
      for (int i = 0; i < 3; ++i) {
        float g = Rn[i * 3 + 0] * x + Rn[i * 3 + 1] * y + Rn[i * 3 + 2] * z + tn[i];
        aPts[(((size_t)(b * Hh + h)) * Nn + n) * 12 + p * 3 + i] = g;
        sq += g * g;
      }
    }
    aSq[((size_t)(b * Hh + h)) * Nn + n] = sq;
  }
  { // b-side (key points)
    const float* Rn = Re + ((size_t)(b * Mm + n)) * 9;
    const float* tn = te + ((size_t)(b * Mm + n)) * 3;
    float sq = 0.f;
    for (int p = 0; p < PQn; ++p) {
      const float* s = kp + ((size_t)(b * Mm + n)) * (Hh * PQn * 3) + (h * PQn + p) * 3;
      float x = s[0], y = s[1], z = s[2];
      for (int i = 0; i < 3; ++i) {
        float g = Rn[i * 3 + 0] * x + Rn[i * 3 + 1] * y + Rn[i * 3 + 2] * z + tn[i];
        bPts[(((size_t)(b * Hh + h)) * Mm + n) * 12 + p * 3 + i] = g;
        sq += g * g;
      }
    }
    bSq[((size_t)(b * Hh + h)) * Mm + n] = sq;
  }
  { // c-side (value points)
    const float* Rn = Re + ((size_t)(b * Mm + n)) * 9;
    const float* tn = te + ((size_t)(b * Mm + n)) * 3;
    for (int p = 0; p < PVn; ++p) {
      const float* s = vp + ((size_t)(b * Mm + n)) * (Hh * PVn * 3) + (h * PVn + p) * 3;
      float x = s[0], y = s[1], z = s[2];
      for (int i = 0; i < 3; ++i) {
        float g = Rn[i * 3 + 0] * x + Rn[i * 3 + 1] * y + Rn[i * 3 + 2] * z + tn[i];
        cPts[(((size_t)(b * Hh + h)) * Mm + n) * 24 + p * 3 + i] = g;
      }
    }
  }
}

// ===========================================================================
// Kernel 3: scores. Block = 16(n) x 32(m) tile, 12 waves (one per head).
//           Pair tile staged ONCE per block via TDM, shared by all heads.
// ===========================================================================
__global__ __launch_bounds__(384)
void scores_kernel(const float* __restrict__ qProj, const float* __restrict__ kProj,
                   const float* __restrict__ aPts,  const float* __restrict__ bPts,
                   const float* __restrict__ aSq,   const float* __restrict__ bSq,
                   const float* __restrict__ pair,  const float* __restrict__ Wpair,
                   const float* __restrict__ head_weight,
                   float* __restrict__ scores)
{
  __shared__ float pr[Cc * 16 * 32];  // 48 KB pair tile, laid out [c][n][m]

  const int tid  = threadIdx.x;
  const int lane = tid & 31;
  const int h    = tid >> 5;            // wave index == head
  const int m16  = lane & 15;
  const int half = lane >> 4;
  const int b    = blockIdx.z;
  const int n0   = blockIdx.y * 16;
  const int m0   = blockIdx.x * 32;

  // TDM: 3D tile x=m(32), y=n(16), z=c(24); stride0=M, stride1=N*M (f32)
  if (tid < 32) {
    tdm_load_to_lds((unsigned)(size_t)(void*)pr,
                    pair + (((size_t)(b * Cc)) * Nn + n0) * Mm + m0,
                    /*dsz_log=*/2, /*tile0=*/32, /*tile1=*/16, /*tile2=*/Cc,
                    (unsigned long long)Mm, (unsigned long long)Nn * Mm);
    __builtin_amdgcn_s_wait_tensorcnt(0);
  }
  __syncthreads();

  // q A-fragments (16 rows x K=64 -> two K=32 chunks)
  v16bf aq[2];
#pragma unroll
  for (int kc = 0; kc < 2; ++kc)
#pragma unroll
    for (int i = 0; i < 8; ++i)
#pragma unroll
      for (int j = 0; j < 2; ++j)
        aq[kc][2 * i + j] = (__bf16)qProj[((size_t)(b * Nn + n0 + m16)) * Dd +
                                          h * DHn + kc * 32 + a_kidx(i, j, half)];
  // point A-fragment (K = 12 zero-padded to 32)
  v16bf ap;
#pragma unroll
  for (int i = 0; i < 8; ++i)
#pragma unroll
    for (int j = 0; j < 2; ++j) {
      int kk = a_kidx(i, j, half);
      ap[2 * i + j] = (kk < 12)
          ? (__bf16)aPts[(((size_t)(b * Hh + h)) * Nn + (n0 + m16)) * 12 + kk]
          : (__bf16)0.0f;
    }

  float wph[Cc];
#pragma unroll
  for (int c = 0; c < Cc; ++c) wph[c] = Wpair[c * Hh + h];

  float asqr[8];
#pragma unroll
  for (int v = 0; v < 8; ++v)
    asqr[v] = aSq[((size_t)(b * Hh + h)) * Nn + n0 + v + 8 * half];

  const float gamma   = log1pf(expf(head_weight[h]));
  const float coef    = 0.5f * sqrtf(2.0f / (9.0f * (float)PQn)) * gamma;
  const float invSqDH = 0.125f;               // 1/sqrt(64)
  const float invSq3  = 0.57735026918962576f;

#pragma unroll
  for (int s = 0; s < 2; ++s) {               // two 16-wide m sub-tiles
    const int mcol = m0 + s * 16 + m16;
    v16bf bq0, bq1, bp;
#pragma unroll
    for (int v = 0; v < 8; ++v)
#pragma unroll
      for (int j = 0; j < 2; ++j) {
        int kk = b_kidx(v, j, half);
        bq0[2 * v + j] = (__bf16)kProj[((size_t)(b * Mm + mcol)) * Dd + h * DHn + kk];
        bq1[2 * v + j] = (__bf16)kProj[((size_t)(b * Mm + mcol)) * Dd + h * DHn + 32 + kk];
        bp[2 * v + j]  = (kk < 12)
            ? (__bf16)bPts[(((size_t)(b * Hh + h)) * Mm + mcol) * 12 + kk]
            : (__bf16)0.0f;
      }
    v8f cqk = {};
    cqk = __builtin_amdgcn_wmma_f32_16x16x32_bf16(false, aq[0], false, bq0, (short)0, cqk, false, false);
    cqk = __builtin_amdgcn_wmma_f32_16x16x32_bf16(false, aq[1], false, bq1, (short)0, cqk, false, false);
    v8f ccr = {};
    ccr = __builtin_amdgcn_wmma_f32_16x16x32_bf16(false, ap, false, bp, (short)0, ccr, false, false);

    const float bsq_m = bSq[((size_t)(b * Hh + h)) * Mm + mcol];
#pragma unroll
    for (int v = 0; v < 8; ++v) {
      const int rloc = v + 8 * half;
      float bias = 0.f;
#pragma unroll
      for (int c = 0; c < Cc; ++c)
        bias += pr[c * 512 + rloc * 32 + s * 16 + m16] * wph[c];
      float inv = asqr[v] + bsq_m - 2.0f * ccr[v];
      float sc  = (cqk[v] * invSqDH + bias - coef * inv) * invSq3;
      scores[(((size_t)(b * Hh + h)) * Nn + (n0 + rloc)) * Mm + mcol] = sc;
    }
  }
}

// ===========================================================================
// Kernel 4: row softmax over m (1024), output bf16 probs
// ===========================================================================
__global__ __launch_bounds__(256)
void softmax_kernel(const float* __restrict__ scores, __bf16* __restrict__ probs)
{
  __shared__ float red[256];
  const size_t row = blockIdx.x;            // (b*H + h)*N + n
  const float* src = scores + row * Mm;
  const int tid = threadIdx.x;

  float vals[4];
  float lm = -1e30f;
#pragma unroll
  for (int i = 0; i < 4; ++i) { vals[i] = src[tid + i * 256]; lm = fmaxf(lm, vals[i]); }
  red[tid] = lm; __syncthreads();
  for (int s = 128; s > 0; s >>= 1) { if (tid < s) red[tid] = fmaxf(red[tid], red[tid + s]); __syncthreads(); }
  const float mx = red[0]; __syncthreads();

  float ls = 0.f;
#pragma unroll
  for (int i = 0; i < 4; ++i) { vals[i] = expf(vals[i] - mx); ls += vals[i]; }
  red[tid] = ls; __syncthreads();
  for (int s = 128; s > 0; s >>= 1) { if (tid < s) red[tid] += red[tid + s]; __syncthreads(); }
  const float inv = 1.0f / red[0];

  __bf16* dst = probs + row * Mm;
#pragma unroll
  for (int i = 0; i < 4; ++i) dst[tid + i * 256] = (__bf16)(vals[i] * inv);
}

// ===========================================================================
// Kernel 5: ctx = probs @ v  and  poG = probs @ cPts  (fused, WMMA)
// ===========================================================================
__global__ __launch_bounds__(128)
void ctx_po_kernel(const __bf16* __restrict__ probs, const float* __restrict__ vProj,
                   const float* __restrict__ cPts, float* __restrict__ ctx,
                   float* __restrict__ poG)
{
  __shared__ float vs[32 * 68];   // v tile    32(K) x 64 (+pad)
  __shared__ float cs[32 * 33];   // cPts tile 32(K) x 32 (+pad), cols>=24 zero

  const int tid  = threadIdx.x;
  const int lane = tid & 31;
  const int wid  = tid >> 5;          // 0..3 -> 16-row strip
  const int m16  = lane & 15;
  const int half = lane >> 4;
  const int bh   = blockIdx.y;
  const int b    = bh / Hh, h = bh % Hh;
  const int n0   = blockIdx.x * 64;

  // staging decomposition (fixed per thread)
  const int vcol = tid & 63, vk0 = tid >> 6;      // vs: kk = vk0 + 2u
  const int ccol = tid & 31, ck0 = tid >> 5;      // cs: kk = ck0 + 4u
  const bool cok = (ccol < 24);

  v8f acc[6] = {};                     // 4 ctx col-subtiles + 2 po col-subtiles

  for (int mt = 0; mt < Mm; mt += 32) {
    const float* gv = vProj + ((size_t)(b * Mm + mt + vk0)) * Dd + h * DHn + vcol;
#pragma unroll
    for (int u = 0; u < 16; ++u)
      vs[(vk0 + 2 * u) * 68 + vcol] = gv[(size_t)(2 * u) * Dd];
    const float* gc = cPts + (((size_t)(b * Hh + h)) * Mm + mt + ck0) * 24 + ccol;
#pragma unroll
    for (int u = 0; u < 8; ++u)
      cs[(ck0 + 4 * u) * 33 + ccol] = cok ? gc[(size_t)(4 * u) * 24] : 0.f;
    __syncthreads();

    v16bf apf;   // probs A-fragment, already bf16 in memory
#pragma unroll
    for (int i = 0; i < 8; ++i)
#pragma unroll
      for (int j = 0; j < 2; ++j)
        apf[2 * i + j] = probs[(((size_t)(b * Hh + h)) * Nn + (n0 + wid * 16 + m16)) * Mm +
                               mt + a_kidx(i, j, half)];

#pragma unroll
    for (int s = 0; s < 4; ++s) {      // ctx cols
      v16bf bf;
#pragma unroll
      for (int v = 0; v < 8; ++v)
#pragma unroll
        for (int j = 0; j < 2; ++j)
          bf[2 * v + j] = (__bf16)vs[b_kidx(v, j, half) * 68 + s * 16 + m16];
      acc[s] = __builtin_amdgcn_wmma_f32_16x16x32_bf16(false, apf, false, bf, (short)0, acc[s], false, false);
    }
#pragma unroll
    for (int s = 0; s < 2; ++s) {      // po cols
      v16bf bf;
#pragma unroll
      for (int v = 0; v < 8; ++v)
#pragma unroll
        for (int j = 0; j < 2; ++j)
          bf[2 * v + j] = (__bf16)cs[b_kidx(v, j, half) * 33 + s * 16 + m16];
      acc[4 + s] = __builtin_amdgcn_wmma_f32_16x16x32_bf16(false, apf, false, bf, (short)0, acc[4 + s], false, false);
    }
    __syncthreads();
  }

#pragma unroll
  for (int v = 0; v < 8; ++v) {
    const int n = n0 + wid * 16 + v + 8 * half;
#pragma unroll
    for (int s = 0; s < 4; ++s)
      ctx[(((size_t)(b * Nn + n)) * Hh + h) * DHn + s * 16 + m16] = acc[s][v];
#pragma unroll
    for (int s = 0; s < 2; ++s) {
      int col = s * 16 + m16;
      if (col < 24)
        poG[(((size_t)(b * Hh + h)) * Nn + n) * 24 + col] = acc[4 + s][v];
    }
  }
}

// ===========================================================================
// Kernel 6: pair aggregation  pair_out[b,n,h,c] = sum_m probs * pair
//           block = 288 threads (9 waves), one (b,n) per block, TDM-staged
// ===========================================================================
__global__ __launch_bounds__(288)
void pair_agg_kernel(const __bf16* __restrict__ probs, const float* __restrict__ pair,
                     float* __restrict__ pair_out)
{
  __shared__ __bf16 plb[Hh * 256];   // probs rows (bf16), [h][m]
  __shared__ float  pp[Cc * 256];    // pair planes,       [c][m]
  const int b = blockIdx.x / Nn, n = blockIdx.x % Nn;
  const int tid = threadIdx.x;
  const int h = tid / Cc, c = tid % Cc;   // block is exactly H*C = 288 threads

  float acc = 0.f;
  for (int mc = 0; mc < Mm; mc += 256) {
    if (tid < 32) {
      // probs: x=m(256) 2B elems, y=h(12) with row stride N*M elements
      tdm_load_to_lds((unsigned)(size_t)(void*)plb,
                      probs + (((size_t)(b * Hh)) * Nn + n) * Mm + mc,
                      /*dsz_log=*/1, /*tile0=*/256, /*tile1=*/Hh, /*tile2=*/0,
                      (unsigned long long)Nn * Mm, 0ull);
      // pair: x=m(256) 4B elems, y=c(24) with plane stride N*M elements
      tdm_load_to_lds((unsigned)(size_t)(void*)pp,
                      pair + (((size_t)(b * Cc)) * Nn + n) * Mm + mc,
                      /*dsz_log=*/2, /*tile0=*/256, /*tile1=*/Cc, /*tile2=*/0,
                      (unsigned long long)Nn * Mm, 0ull);
      __builtin_amdgcn_s_wait_tensorcnt(0);
    }
    __syncthreads();
#pragma unroll 8
    for (int m = 0; m < 256; ++m) acc += (float)plb[h * 256 + m] * pp[c * 256 + m];
    __syncthreads();
  }
  pair_out[(size_t)blockIdx.x * (Hh * Cc) + tid] = acc;
}

// ===========================================================================
// Kernel 7: finalize — back-rotate po with R^T, norms, concat features
// ===========================================================================
__global__ __launch_bounds__(128)
void finalize_feats_kernel(const float* __restrict__ pair_out, const float* __restrict__ ctx,
                           const float* __restrict__ poG, const float* __restrict__ R,
                           const float* __restrict__ t, float* __restrict__ feats)
{
  const int bn = blockIdx.x;
  const int b = bn / Nn, n = bn % Nn;
  float* f = feats + (size_t)bn * FEAT;
  const int tid = threadIdx.x;

  __shared__ float Rn[9];
  __shared__ float tn[3];
  if (tid < 9) Rn[tid] = R[(size_t)bn * 9 + tid];
  if (tid < 3) tn[tid] = t[(size_t)bn * 3 + tid];

  for (int i = tid; i < Hh * Cc; i += 128) f[i] = pair_out[(size_t)bn * (Hh * Cc) + i];
  for (int i = tid; i < Hh * DHn; i += 128) f[Hh * Cc + i] = ctx[(size_t)bn * (Hh * DHn) + i];
  __syncthreads();

  for (int i = tid; i < Hh * PVn; i += 128) {
    int h = i / PVn, p = i % PVn;
    const float* g = poG + (((size_t)(b * Hh + h)) * Nn + n) * 24 + p * 3;
    float gx = g[0] - tn[0], gy = g[1] - tn[1], gz = g[2] - tn[2];
    float l0 = Rn[0] * gx + Rn[3] * gy + Rn[6] * gz;   // R^T
    float l1 = Rn[1] * gx + Rn[4] * gy + Rn[7] * gz;
    float l2 = Rn[2] * gx + Rn[5] * gy + Rn[8] * gz;
    const int base = Hh * Cc + Hh * DHn;               // 288 + 768
    f[base + i * 3 + 0] = l0;
    f[base + i * 3 + 1] = l1;
    f[base + i * 3 + 2] = l2;
    f[base + Hh * PVn * 3 + i] = sqrtf(l0 * l0 + l1 * l1 + l2 * l2);
  }
}

// ===========================================================================
// Host-side launch
// ===========================================================================
extern "C" void kernel_launch(void* const* d_in, const int* in_sizes, int n_in,
                              void* d_out, int out_size, void* d_ws, size_t ws_size,
                              hipStream_t stream)
{
  (void)in_sizes; (void)n_in; (void)out_size; (void)ws_size;

  const float* hidden  = (const float*)d_in[0];
  const float* encoder = (const float*)d_in[1];
  const float* pair    = (const float*)d_in[2];
  const float* R       = (const float*)d_in[3];
  const float* t       = (const float*)d_in[4];
  const float* Re      = (const float*)d_in[5];
  const float* te      = (const float*)d_in[6];
  const float* Wq      = (const float*)d_in[7];
  const float* Wk      = (const float*)d_in[8];
  const float* Wv      = (const float*)d_in[9];
  const float* Wqp     = (const float*)d_in[10];
  const float* Wkp     = (const float*)d_in[11];
  const float* Wvp     = (const float*)d_in[12];
  const float* hw      = (const float*)d_in[13];
  const float* Wpair   = (const float*)d_in[14];
  const float* Wout    = (const float*)d_in[15];
  const float* bout    = (const float*)d_in[16];
  float* out = (float*)d_out;

  float*  ws    = (float*)d_ws;
  __bf16* probs = (__bf16*)(ws + OFF_PROBS);

  const int rows = Bq * Nn;  // 2048

  // 1) scalar + point projections
  {
    dim3 blk(256);
    dim3 gD((Dd + 63) / 64, rows / 64);
    hipLaunchKernelGGL(gemm_bf16_kernel, gD, blk, 0, stream, hidden,  Wq,  nullptr, ws + OFF_Q, rows, Dd, Dd);
    hipLaunchKernelGGL(gemm_bf16_kernel, gD, blk, 0, stream, encoder, Wk,  nullptr, ws + OFF_K, rows, Dd, Dd);
    hipLaunchKernelGGL(gemm_bf16_kernel, gD, blk, 0, stream, encoder, Wv,  nullptr, ws + OFF_V, rows, Dd, Dd);
    dim3 gP((Hh * PQn * 3 + 63) / 64, rows / 64);
    hipLaunchKernelGGL(gemm_bf16_kernel, gP, blk, 0, stream, hidden,  Wqp, nullptr, ws + OFF_QP, rows, Dd, Hh * PQn * 3);
    hipLaunchKernelGGL(gemm_bf16_kernel, gP, blk, 0, stream, encoder, Wkp, nullptr, ws + OFF_KP, rows, Dd, Hh * PQn * 3);
    dim3 gV((Hh * PVn * 3 + 63) / 64, rows / 64);
    hipLaunchKernelGGL(gemm_bf16_kernel, gV, blk, 0, stream, encoder, Wvp, nullptr, ws + OFF_VP, rows, Dd, Hh * PVn * 3);
  }

  // 2) rotate points to global frame
  {
    int total = Bq * Nn * Hh;
    hipLaunchKernelGGL(rotate_points_kernel, dim3((total + 255) / 256), dim3(256), 0, stream,
                       ws + OFF_QP, ws + OFF_KP, ws + OFF_VP, R, t, Re, te,
                       ws + OFF_APTS, ws + OFF_BPTS, ws + OFF_CPTS,
                       ws + OFF_ASQ, ws + OFF_BSQ);
  }

  // 3) scores (QK^T + pair bias + point invariant), TDM pair staging
  hipLaunchKernelGGL(scores_kernel, dim3(Mm / 32, Nn / 16, Bq), dim3(384), 0, stream,
                     ws + OFF_Q, ws + OFF_K, ws + OFF_APTS, ws + OFF_BPTS,
                     ws + OFF_ASQ, ws + OFF_BSQ, pair, Wpair, hw, ws + OFF_SC);

  // 4) softmax -> bf16 probs
  hipLaunchKernelGGL(softmax_kernel, dim3(Bq * Hh * Nn), dim3(256), 0, stream,
                     ws + OFF_SC, probs);

  // 5) ctx + global-frame point aggregation
  hipLaunchKernelGGL(ctx_po_kernel, dim3(Nn / 64, Bq * Hh), dim3(128), 0, stream,
                     probs, ws + OFF_V, ws + OFF_CPTS, ws + OFF_CTX, ws + OFF_POG);

  // 6) pair aggregation (TDM staged)
  hipLaunchKernelGGL(pair_agg_kernel, dim3(Bq * Nn), dim3(288), 0, stream,
                     probs, pair, ws + OFF_POUT);

  // 7) finalize features
  hipLaunchKernelGGL(finalize_feats_kernel, dim3(Bq * Nn), dim3(128), 0, stream,
                     ws + OFF_POUT, ws + OFF_CTX, ws + OFF_POG, R, t, ws + OFF_FE);

  // 8) output projection with bias
  hipLaunchKernelGGL(gemm_bf16_kernel, dim3(Dd / 64, rows / 64), dim3(256), 0, stream,
                     ws + OFF_FE, Wout, bout, out, rows, FEAT, Dd);
}
